// PhaseAwareAttention_39462159515742
// MI455X (gfx1250) — compile-verified, matrix operand order checked by reading, not run on hardware
//
#include <hip/hip_runtime.h>
#include <hip/hip_bf16.h>

// ---------------------------------------------------------------------------
// PhaseAwareAttention for MI455X (gfx1250), wave32 + WMMA bf16 path.
//   k0: fp32 -> bf16 pre-conversion of x / qkv_w / proj_w (one pass)
//   k1: qkv GEMM    (bf16 wmma, 16x64 tile per wave, double-buffered loads)
//   k2: flash attn  (bf16 wmma, online softmax, LDS P-transpose, pipelined K)
//   k3: proj GEMM   (bf16 wmma, double-buffered loads)
// Workspace layout (assumes ws_size >= 48 MiB):
//   [ 0, 8M)  xb   bf16 [B*N, C]
//   [ 8,14M)  wqkv bf16 [3C, C]
//   [14,16M)  wprj bf16 [C, C]
//   [16,24M)  q    bf16 [B,H,N,D]
//   [24,32M)  k    bf16 [B,H,N,D]
//   [32,40M)  vT   bf16 [B,H,D,N]
//   [40,48M)  ao   bf16 [B*N, C]
// ---------------------------------------------------------------------------

typedef __attribute__((ext_vector_type(16))) __bf16 v16bf;
typedef __attribute__((ext_vector_type(8)))  __bf16 v8bf;
typedef __attribute__((ext_vector_type(8)))  float  v8f;

#define EMBED 1024
#define HEADS 16
#define HDIM  64
#define BATCH 2
#define SEQ   2048

__device__ __forceinline__ v8f wmma_bf16(v16bf a, v16bf b, v8f c) {
  // (neg_a, A, neg_b, B, c_mod, C, reuse_a, reuse_b)
  return __builtin_amdgcn_wmma_f32_16x16x32_bf16(false, a, false, b,
                                                 (short)0, c, false, false);
}

__device__ __forceinline__ v16bf cat16(v8bf lo, v8bf hi) {
  v16bf a;
#pragma unroll
  for (int i = 0; i < 8; ++i) { a[i] = lo[i]; a[i + 8] = hi[i]; }
  return a;
}

__device__ __forceinline__ v8bf ld8(const __bf16* p) { return *(const v8bf*)p; }

// ---------------------------------------------------------------------------
// Kernel 0: fp32 -> bf16 bulk conversion (8 elements / thread).
// ---------------------------------------------------------------------------
__global__ __launch_bounds__(256) void cvt_bf16_kernel(
    const float* __restrict__ src, __bf16* __restrict__ dst, int n) {
  const int i = (blockIdx.x * 256 + threadIdx.x) * 8;
  if (i < n) {
    float4 f0 = *(const float4*)(src + i);
    float4 f1 = *(const float4*)(src + i + 4);
    v8bf o;
    o[0] = (__bf16)f0.x; o[1] = (__bf16)f0.y; o[2] = (__bf16)f0.z; o[3] = (__bf16)f0.w;
    o[4] = (__bf16)f1.x; o[5] = (__bf16)f1.y; o[6] = (__bf16)f1.z; o[7] = (__bf16)f1.w;
    *(v8bf*)(dst + i) = o;
  }
}

// ---------------------------------------------------------------------------
// Shared GEMM machinery: 16(M) x 64(N) tile per wave over K=1024.
// Fragment set for one k-step: A lo/hi + 4 x B lo/hi = 10 x v8bf (40 VGPRs).
// ---------------------------------------------------------------------------
struct Frag10 { v8bf d[10]; };

__device__ __forceinline__ void load_tile10(
    Frag10& f, const __bf16* arow, const __bf16* b0, const __bf16* b1,
    const __bf16* b2, const __bf16* b3, int kk) {
  f.d[0] = ld8(arow + kk); f.d[1] = ld8(arow + kk + 16);
  f.d[2] = ld8(b0 + kk);   f.d[3] = ld8(b0 + kk + 16);
  f.d[4] = ld8(b1 + kk);   f.d[5] = ld8(b1 + kk + 16);
  f.d[6] = ld8(b2 + kk);   f.d[7] = ld8(b2 + kk + 16);
  f.d[8] = ld8(b3 + kk);   f.d[9] = ld8(b3 + kk + 16);
}

__device__ __forceinline__ void wmma_step(v8f acc[4], const Frag10& f) {
  const v16bf a = cat16(f.d[0], f.d[1]);
  acc[0] = wmma_bf16(a, cat16(f.d[2], f.d[3]), acc[0]);
  acc[1] = wmma_bf16(a, cat16(f.d[4], f.d[5]), acc[1]);
  acc[2] = wmma_bf16(a, cat16(f.d[6], f.d[7]), acc[2]);
  acc[3] = wmma_bf16(a, cat16(f.d[8], f.d[9]), acc[3]);
}

// ---------------------------------------------------------------------------
// Kernel 1: QKV projection.  out[m][n] = sum_k x[m][k] * qkv_w[n][k] + b[n]
// Software-pipelined: k-step kk+32 loads in flight while kk is consumed.
// ---------------------------------------------------------------------------
__global__ __launch_bounds__(32) void qkv_kernel(
    const __bf16* __restrict__ xb, const __bf16* __restrict__ wb,
    const float* __restrict__ bias,
    __bf16* __restrict__ qws, __bf16* __restrict__ kws, __bf16* __restrict__ vTws) {
  const int lane  = threadIdx.x;
  const int mlane = lane & 15;
  const int half  = lane >> 4;
  const int m0    = blockIdx.x * 16;
  const int n0    = blockIdx.y * 64;

  const __bf16* arow  = xb + (size_t)(m0 + mlane) * EMBED + half * 8;
  const __bf16* brow0 = wb + (size_t)(n0 +  0 + mlane) * EMBED + half * 8;
  const __bf16* brow1 = wb + (size_t)(n0 + 16 + mlane) * EMBED + half * 8;
  const __bf16* brow2 = wb + (size_t)(n0 + 32 + mlane) * EMBED + half * 8;
  const __bf16* brow3 = wb + (size_t)(n0 + 48 + mlane) * EMBED + half * 8;

  v8f acc[4] = {v8f{}, v8f{}, v8f{}, v8f{}};

  Frag10 cur, nxt;
  load_tile10(cur, arow, brow0, brow1, brow2, brow3, 0);
  for (int kk = 0; kk < EMBED - 32; kk += 32) {
    load_tile10(nxt, arow, brow0, brow1, brow2, brow3, kk + 32);
    wmma_step(acc, cur);
    cur = nxt;
  }
  wmma_step(acc, cur);

  // Scatter-store: C/D layout: VGPR r -> row (m0 + half*8 + r), lane&15 -> col.
#pragma unroll
  for (int t = 0; t < 4; ++t) {
    const int n = n0 + t * 16 + mlane;
    const float bv = bias[n];
    const int qkv_idx = n >> 10;          // uniform across the wave
    const int rem = n & 1023;
    const int h = rem >> 6;
    const int d = rem & 63;
#pragma unroll
    for (int r = 0; r < 8; ++r) {
      const int mrow = m0 + half * 8 + r;
      const int bb = mrow >> 11;          // batch
      const int tok = mrow & 2047;        // token
      const __bf16 val = (__bf16)(acc[t][r] + bv);
      if (qkv_idx == 0) {
        qws[(((size_t)bb * HEADS + h) * SEQ + tok) * HDIM + d] = val;
      } else if (qkv_idx == 1) {
        kws[(((size_t)bb * HEADS + h) * SEQ + tok) * HDIM + d] = val;
      } else {
        vTws[(((size_t)bb * HEADS + h) * HDIM + d) * SEQ + tok] = val;
      }
    }
  }
}

// ---------------------------------------------------------------------------
// Kernel 2: flash attention, one wave per (b, h, 16-query tile).
// K fragments pipelined one 32-key tile ahead; V loads hidden under softmax.
// ---------------------------------------------------------------------------
struct KFrags { v8bf d[8]; };

__device__ __forceinline__ void load_kfrags(
    KFrags& f, const __bf16* krow0, const __bf16* krow1, int koff) {
  f.d[0] = ld8(krow0 + koff);      f.d[1] = ld8(krow0 + koff + 16);
  f.d[2] = ld8(krow0 + koff + 32); f.d[3] = ld8(krow0 + koff + 48);
  f.d[4] = ld8(krow1 + koff);      f.d[5] = ld8(krow1 + koff + 16);
  f.d[6] = ld8(krow1 + koff + 32); f.d[7] = ld8(krow1 + koff + 48);
}

__global__ __launch_bounds__(32) void attn_kernel(
    const __bf16* __restrict__ qb, const __bf16* __restrict__ kb,
    const __bf16* __restrict__ vTb, __bf16* __restrict__ ao,
    const float* __restrict__ phase_signal, const float* __restrict__ phase_weight) {
  const int lane = threadIdx.x;
  const int m    = lane & 15;
  const int half = lane >> 4;
  const int bh   = blockIdx.x;
  const int bb   = bh >> 4;
  const int h    = bh & 15;
  const int q0   = blockIdx.y * 16;

  // D^-0.5 * (1 + 0.2 * phase_weight[h] * phase_signal[b]) folded into scale
  const float scale = 0.125f * (1.0f + 0.2f * phase_weight[h] * phase_signal[bb]);

  const size_t headQ = ((size_t)bb * HEADS + h) * SEQ;   // row base for q/k
  const size_t headV = ((size_t)bb * HEADS + h) * HDIM;  // row base for vT

  // Q fragments (reused for all key tiles): K-dim = head_dim, two 32-chunks.
  const __bf16* qrow = qb + (headQ + q0 + m) * HDIM + half * 8;
  const v16bf qa0 = cat16(ld8(qrow), ld8(qrow + 16));
  const v16bf qa1 = cat16(ld8(qrow + 32), ld8(qrow + 48));

  // Fixed per-lane row bases for K and vT streams.
  const __bf16* krow0 = kb + (headQ + m) * HDIM + half * 8;        // keys j..j+15
  const __bf16* krow1 = kb + (headQ + 16 + m) * HDIM + half * 8;   // keys j+16..j+31
  const __bf16* vrow0 = vTb + (headV +  0 + m) * SEQ + half * 8;
  const __bf16* vrow1 = vTb + (headV + 16 + m) * SEQ + half * 8;
  const __bf16* vrow2 = vTb + (headV + 32 + m) * SEQ + half * 8;
  const __bf16* vrow3 = vTb + (headV + 48 + m) * SEQ + half * 8;

  v8f o0 = {}, o1 = {}, o2 = {}, o3 = {};
  float mrow[8], lrow[8];
#pragma unroll
  for (int r = 0; r < 8; ++r) { mrow[r] = -3.0e38f; lrow[r] = 0.0f; }

  __shared__ __bf16 lds_p[32 * 16];  // P tile staging (16 rows x 32 keys, bf16)

  KFrags kcur, knxt;
  load_kfrags(kcur, krow0, krow1, 0);

  for (int j = 0; j < SEQ; j += 32) {
    // ---- V loads for this tile: latency hides under softmax VALU below ----
    v8bf v0l = ld8(vrow0 + j), v0h = ld8(vrow0 + j + 16);
    v8bf v1l = ld8(vrow1 + j), v1h = ld8(vrow1 + j + 16);
    v8bf v2l = ld8(vrow2 + j), v2h = ld8(vrow2 + j + 16);
    v8bf v3l = ld8(vrow3 + j), v3h = ld8(vrow3 + j + 16);

    // ---- S = (Q K^T) * scale from the pipelined K fragments ----
    v8f s0 = {}, s1 = {};
    s0 = wmma_bf16(qa0, cat16(kcur.d[0], kcur.d[1]), s0);
    s0 = wmma_bf16(qa1, cat16(kcur.d[2], kcur.d[3]), s0);
    s1 = wmma_bf16(qa0, cat16(kcur.d[4], kcur.d[5]), s1);
    s1 = wmma_bf16(qa1, cat16(kcur.d[6], kcur.d[7]), s1);

    // ---- issue next tile's K loads now; they fly during softmax+LDS ----
    const int jn = (j + 32 < SEQ) ? (j + 32) : j;  // clamped, branch-free
    load_kfrags(knxt, krow0, krow1, jn * HDIM);
    __builtin_prefetch(vrow0 + jn, 0, 0);          // global_prefetch_b8 hint

    // ---- online softmax update (row reductions across 16-lane halves) ----
    float p0[8], p1[8], alpha[8];
#pragma unroll
    for (int r = 0; r < 8; ++r) {
      const float a = s0[r] * scale;
      const float c = s1[r] * scale;
      float mt = fmaxf(a, c);
      mt = fmaxf(mt, __shfl_xor(mt, 1, 16));
      mt = fmaxf(mt, __shfl_xor(mt, 2, 16));
      mt = fmaxf(mt, __shfl_xor(mt, 4, 16));
      mt = fmaxf(mt, __shfl_xor(mt, 8, 16));
      const float mn = fmaxf(mrow[r], mt);
      alpha[r] = __expf(mrow[r] - mn);
      mrow[r] = mn;
      p0[r] = __expf(a - mn);
      p1[r] = __expf(c - mn);
      float rs = p0[r] + p1[r];
      rs += __shfl_xor(rs, 1, 16);
      rs += __shfl_xor(rs, 2, 16);
      rs += __shfl_xor(rs, 4, 16);
      rs += __shfl_xor(rs, 8, 16);
      lrow[r] = lrow[r] * alpha[r] + rs;
    }
#pragma unroll
    for (int r = 0; r < 8; ++r) {
      o0[r] *= alpha[r]; o1[r] *= alpha[r]; o2[r] *= alpha[r]; o3[r] *= alpha[r];
    }

    // ---- C/D-layout P -> A-layout bf16 fragment via LDS round trip ----
    __syncthreads();
#pragma unroll
    for (int r = 0; r < 8; ++r) {
      lds_p[lane * 16 + r]     = (__bf16)p0[r];  // local cols 0..15
      lds_p[lane * 16 + 8 + r] = (__bf16)p1[r];  // local cols 16..31
    }
    __syncthreads();
    v16bf pa;
    {
      const int rsel  = m & 7;
      const int wbase = (m >> 3) * 16;  // rows 8..15 were written by lanes 16..31
      const int kcq   = half * 8;       // A-layout K chunk for this lane
#pragma unroll
      for (int i = 0; i < 8; ++i) {
        const int wl = wbase + kcq + i;
        pa[i]     = lds_p[wl * 16 + rsel];      // K = kcq+i
        pa[i + 8] = lds_p[wl * 16 + 8 + rsel];  // K = kcq+16+i
      }
    }
    __syncthreads();

    // ---- O += P @ V ----
    o0 = wmma_bf16(pa, cat16(v0l, v0h), o0);
    o1 = wmma_bf16(pa, cat16(v1l, v1h), o1);
    o2 = wmma_bf16(pa, cat16(v2l, v2h), o2);
    o3 = wmma_bf16(pa, cat16(v3l, v3h), o3);

    kcur = knxt;
  }

  // ---- finalize (divide by softmax denom) and store bf16 [B*N, C] ----
#pragma unroll
  for (int r = 0; r < 8; ++r) {
    const float inv = 1.0f / lrow[r];
    o0[r] *= inv; o1[r] *= inv; o2[r] *= inv; o3[r] *= inv;
  }
#pragma unroll
  for (int r = 0; r < 8; ++r) {
    const int tok = q0 + half * 8 + r;
    const size_t base = ((size_t)bb * SEQ + tok) * EMBED + h * HDIM + m;
    ao[base + 0]  = (__bf16)o0[r];
    ao[base + 16] = (__bf16)o1[r];
    ao[base + 32] = (__bf16)o2[r];
    ao[base + 48] = (__bf16)o3[r];
  }
}

// ---------------------------------------------------------------------------
// Kernel 3: output projection.  out[m][n] = sum_k ao[m][k]*proj_w[n][k] + b[n]
// ---------------------------------------------------------------------------
__global__ __launch_bounds__(32) void proj_kernel(
    const __bf16* __restrict__ ain, const __bf16* __restrict__ wb,
    const float* __restrict__ bias, float* __restrict__ out) {
  const int lane  = threadIdx.x;
  const int mlane = lane & 15;
  const int half  = lane >> 4;
  const int m0    = blockIdx.x * 16;
  const int n0    = blockIdx.y * 64;

  const __bf16* arow  = ain + (size_t)(m0 + mlane) * EMBED + half * 8;
  const __bf16* brow0 = wb + (size_t)(n0 +  0 + mlane) * EMBED + half * 8;
  const __bf16* brow1 = wb + (size_t)(n0 + 16 + mlane) * EMBED + half * 8;
  const __bf16* brow2 = wb + (size_t)(n0 + 32 + mlane) * EMBED + half * 8;
  const __bf16* brow3 = wb + (size_t)(n0 + 48 + mlane) * EMBED + half * 8;

  v8f acc[4] = {v8f{}, v8f{}, v8f{}, v8f{}};

  Frag10 cur, nxt;
  load_tile10(cur, arow, brow0, brow1, brow2, brow3, 0);
  for (int kk = 0; kk < EMBED - 32; kk += 32) {
    load_tile10(nxt, arow, brow0, brow1, brow2, brow3, kk + 32);
    wmma_step(acc, cur);
    cur = nxt;
  }
  wmma_step(acc, cur);

#pragma unroll
  for (int t = 0; t < 4; ++t) {
    const int n = n0 + t * 16 + mlane;
    const float bv = bias[n];
#pragma unroll
    for (int r = 0; r < 8; ++r) {
      const int mrow = m0 + half * 8 + r;
      out[(size_t)mrow * EMBED + n] = acc[t][r] + bv;
    }
  }
}

// ---------------------------------------------------------------------------
extern "C" void kernel_launch(void* const* d_in, const int* in_sizes, int n_in,
                              void* d_out, int out_size, void* d_ws, size_t ws_size,
                              hipStream_t stream) {
  const float* x            = (const float*)d_in[0];
  const float* phase_signal = (const float*)d_in[1];
  const float* qkv_w        = (const float*)d_in[2];
  const float* qkv_b        = (const float*)d_in[3];
  const float* proj_w       = (const float*)d_in[4];
  const float* proj_b       = (const float*)d_in[5];
  const float* phase_weight = (const float*)d_in[6];
  float* out = (float*)d_out;

  char* ws = (char*)d_ws;
  const size_t MB = 1024 * 1024;
  __bf16* xb    = (__bf16*)(ws + 0 * MB);
  __bf16* wqkvb = (__bf16*)(ws + 8 * MB);
  __bf16* wprjb = (__bf16*)(ws + 14 * MB);
  __bf16* qws   = (__bf16*)(ws + 16 * MB);
  __bf16* kws   = (__bf16*)(ws + 24 * MB);
  __bf16* vT    = (__bf16*)(ws + 32 * MB);
  __bf16* ao    = (__bf16*)(ws + 40 * MB);

  // k0: one-pass fp32 -> bf16 conversions (8 elems/thread, 256 threads/block)
  cvt_bf16_kernel<<<dim3((BATCH * SEQ * EMBED) / 8 / 256), 256, 0, stream>>>(
      x, xb, BATCH * SEQ * EMBED);
  cvt_bf16_kernel<<<dim3((3 * EMBED * EMBED) / 8 / 256), 256, 0, stream>>>(
      qkv_w, wqkvb, 3 * EMBED * EMBED);
  cvt_bf16_kernel<<<dim3((EMBED * EMBED) / 8 / 256), 256, 0, stream>>>(
      proj_w, wprjb, EMBED * EMBED);

  // k1: QKV GEMM: M=4096 -> 256 row tiles; N=3072 -> 48 col tiles of 64
  qkv_kernel<<<dim3(256, 48), 32, 0, stream>>>(xb, wqkvb, qkv_b, qws, kws, vT);
  // k2: attention: 32 (b,h) pairs x 128 query tiles of 16
  attn_kernel<<<dim3(32, 128), 32, 0, stream>>>(qws, kws, vT, ao,
                                                phase_signal, phase_weight);
  // k3: projection: 256 row tiles x 16 col tiles of 64
  proj_kernel<<<dim3(256, 16), 32, 0, stream>>>(ao, wprjb, proj_b, out);
}
